// UnderstanderDecoder_35313221107944
// MI455X (gfx1250) — matrix operand order repaired and checked by reading, not compiled, not run on hardware
//
#include <hip/hip_runtime.h>
#include <math.h>

// Problem constants from the reference (B, E, H, DEC) = (16, 1024, 1024, 64).
#define B_DIM     16
#define E_DIM     1024
#define H_DIM     1024
#define DEC_STEPS 64
#define G_DIM     (4 * H_DIM)

typedef __attribute__((ext_vector_type(16))) __bf16 v16bf;
typedef __attribute__((ext_vector_type(8)))  __bf16 v8bf;
typedef __attribute__((ext_vector_type(4)))  __bf16 v4bf;
typedef __attribute__((ext_vector_type(8)))  float  v8f;

// D = A*B + C on the bf16 XDL path (wave32, 16x16x32).
#define WMMA_BF16(acc, a, b)                                                   \
  (acc) = __builtin_amdgcn_wmma_f32_16x16x32_bf16(false, (a), false, (b),      \
                                                  (short)0, (acc), false, false)

// ---------------------------------------------------------------------------
// Fragment loaders from PRE-SPLIT bf16 arrays (plain row-major) — pure loads.
// Layouts per CDNA5 ISA 7.12.2 (wave32):
//   A 16x32 bf16 : lanes 0-15 -> M=lane, elems {K 0..7, 16..23};
//                  lanes 16-31 -> M=lane-16, elems {K 8..15, 24..31}
//   B 32x16 bf16 : lanes 0-15 -> N=lane, elems {K 0..7, 16..23};
//                  lanes 16-31 -> N=lane-16, elems {K 8..15, 24..31}
//   C/D 16x16 f32: VGPR r -> M = r (+8 for lanes 16-31), N = lane&15
// ---------------------------------------------------------------------------
__device__ __forceinline__ v16bf load_a_frag_bf16(const __bf16* __restrict__ A,
                                                  int lda, int k0, int lane) {
  const int m  = lane & 15;
  const int kb = (lane & 16) ? 8 : 0;
  const __bf16* p = A + (size_t)m * lda + k0 + kb;
  const v8bf x0 = *(const v8bf*)(p);        // K kb+0..7   (16B aligned)
  const v8bf x1 = *(const v8bf*)(p + 16);   // K kb+16..23 (16B aligned)
  return __builtin_shufflevector(x0, x1, 0, 1, 2, 3, 4, 5, 6, 7,
                                 8, 9, 10, 11, 12, 13, 14, 15);
}

__device__ __forceinline__ v16bf load_b_frag_bf16(const __bf16* __restrict__ W,
                                                  int ldw, int n0, int col, int lane) {
  const int n   = lane & 15;
  const int kkb = (lane & 16) ? 8 : 0;
  const __bf16* p = W + (size_t)(n0 + n) * ldw + col + kkb;
  const v8bf y0 = *(const v8bf*)(p);        // K kkb+0..7
  const v8bf y1 = *(const v8bf*)(p + 16);   // K kkb+16..23
  return __builtin_shufflevector(y0, y1, 0, 1, 2, 3, 4, 5, 6, 7,
                                 8, 9, 10, 11, 12, 13, 14, 15);
}

// Out(16 x NT*16) = A(16xK) * W(n,k)^T with error-compensated bf16:
// (ah+al)(bh+bl) ~= ah*bh + ah*bl + al*bh, fp32 accumulate.
// One A fragment feeds NT tiles (12 WMMAs per A load when NT=4).
template <int NT>
__device__ __forceinline__ void gemm_rowtile(
    const __bf16* __restrict__ Ahi, const __bf16* __restrict__ Alo, int lda,
    const __bf16* __restrict__ Whi, const __bf16* __restrict__ Wlo, int ldw,
    int n0, int woff, int K, int lane, v8f acc[NT]) {
#pragma unroll
  for (int t = 0; t < NT; ++t) acc[t] = (v8f){0.f, 0.f, 0.f, 0.f, 0.f, 0.f, 0.f, 0.f};
  for (int k0 = 0; k0 < K; k0 += 32) {
    if (k0 + 32 < K) {  // gfx1250 global_prefetch_b8 on next A tile
      __builtin_prefetch(Ahi + (size_t)(lane & 15) * lda + k0 + 32, 0, 2);
      __builtin_prefetch(Alo + (size_t)(lane & 15) * lda + k0 + 32, 0, 2);
    }
    const v16bf ah = load_a_frag_bf16(Ahi, lda, k0, lane);
    const v16bf al = load_a_frag_bf16(Alo, lda, k0, lane);
#pragma unroll
    for (int t = 0; t < NT; ++t) {
      const v16bf bh = load_b_frag_bf16(Whi, ldw, n0 + t * 16, woff + k0, lane);
      const v16bf bl = load_b_frag_bf16(Wlo, ldw, n0 + t * 16, woff + k0, lane);
      WMMA_BF16(acc[t], ah, bh);
      WMMA_BF16(acc[t], ah, bl);
      WMMA_BF16(acc[t], al, bh);
    }
  }
}

// ---------------------------------------------------------------------------
// Prep: split f32 -> bf16 hi/lo pair (x ~= hi + lo). 4 elements/thread.
// ---------------------------------------------------------------------------
__global__ __launch_bounds__(256) void split_bf16_kernel(
    const float* __restrict__ src, __bf16* __restrict__ hi,
    __bf16* __restrict__ lo) {
  const int i = blockIdx.x * blockDim.x + threadIdx.x;
  const float4 x = ((const float4*)src)[i];
  v4bf h, l;
  h[0] = (__bf16)x.x; l[0] = (__bf16)(x.x - (float)h[0]);
  h[1] = (__bf16)x.y; l[1] = (__bf16)(x.y - (float)h[1]);
  h[2] = (__bf16)x.z; l[2] = (__bf16)(x.z - (float)h[2]);
  h[3] = (__bf16)x.w; l[3] = (__bf16)(x.w - (float)h[3]);
  ((v4bf*)hi)[i] = h;
  ((v4bf*)lo)[i] = l;
}

// ---------------------------------------------------------------------------
// Kernel 1: enc_part[be, k] = sum_h enc[be, h] * W_hid[k, h] + b_hid[k]
// 16 N-tiles per block (4 waves x 4 tiles); bf16 operands pre-split.
// ---------------------------------------------------------------------------
__global__ __launch_bounds__(128) void enc_gemm_kernel(
    const __bf16* __restrict__ enc_hi, const __bf16* __restrict__ enc_lo,
    const __bf16* __restrict__ whid_hi, const __bf16* __restrict__ whid_lo,
    const float* __restrict__ b_hid, float* __restrict__ enc_part) {
  const int lane = threadIdx.x & 31;
  const int wave = threadIdx.x >> 5;
  const int m0 = blockIdx.x * 16;                        // tile over B*E rows
  const int n0 = (blockIdx.y * 16 + wave * 4) * 16;      // 4 tiles over H cols

  v8f acc[4];
  gemm_rowtile<4>(enc_hi + (size_t)m0 * H_DIM, enc_lo + (size_t)m0 * H_DIM, H_DIM,
                  whid_hi, whid_lo, 2 * H_DIM, n0, /*woff=*/0, H_DIM, lane, acc);

  const int n  = lane & 15;
  const int mb = (lane & 16) ? 8 : 0;
#pragma unroll
  for (int t = 0; t < 4; ++t) {
    const float bias = b_hid[n0 + t * 16 + n];
    float* out = enc_part + (size_t)(m0 + mb) * H_DIM + n0 + t * 16 + n;
#pragma unroll
    for (int r = 0; r < 8; ++r) out[(size_t)r * H_DIM] = acc[t][r] + bias;
  }
}

// ---------------------------------------------------------------------------
// Kernel 2: gates[b, j] = sum_h h[b,h] * W_hh[j,h] + (b_ih[j] + b_hh[j])
// ---------------------------------------------------------------------------
__global__ __launch_bounds__(128) void gates_gemm_kernel(
    const __bf16* __restrict__ h_hi, const __bf16* __restrict__ h_lo,
    const __bf16* __restrict__ whh_hi, const __bf16* __restrict__ whh_lo,
    const float* __restrict__ b_ih, const float* __restrict__ b_hh,
    float* __restrict__ gates) {
  const int lane = threadIdx.x & 31;
  const int wave = threadIdx.x >> 5;
  const int n0 = (blockIdx.x * 16 + wave * 4) * 16;      // 4 of 256 tiles

  v8f acc[4];
  gemm_rowtile<4>(h_hi, h_lo, H_DIM, whh_hi, whh_lo, H_DIM, n0, 0, H_DIM, lane, acc);

  const int n  = lane & 15;
  const int mb = (lane & 16) ? 8 : 0;
#pragma unroll
  for (int t = 0; t < 4; ++t) {
    const float bias = b_ih[n0 + t * 16 + n] + b_hh[n0 + t * 16 + n];
    float* out = gates + (size_t)mb * G_DIM + n0 + t * 16 + n;
#pragma unroll
    for (int r = 0; r < 8; ++r) out[(size_t)r * G_DIM] = acc[t][r] + bias;
  }
}

// ---------------------------------------------------------------------------
// Kernel 3: LSTM pointwise; emits next h directly as split bf16 hi/lo.
// ---------------------------------------------------------------------------
__global__ __launch_bounds__(256) void lstm_cell_kernel(
    const float* __restrict__ gates, float* __restrict__ cbuf,
    __bf16* __restrict__ h_hi, __bf16* __restrict__ h_lo) {
  const int idx = blockIdx.x * blockDim.x + threadIdx.x; // < 16384
  const int b = idx >> 10, k = idx & (H_DIM - 1);
  const float* g = gates + (size_t)b * G_DIM;
  const float gi = g[k];
  const float gf = g[H_DIM + k];
  const float gg = g[2 * H_DIM + k];
  const float go = g[3 * H_DIM + k];
  const float si = 1.f / (1.f + __expf(-gi));
  const float sf = 1.f / (1.f + __expf(-gf));
  const float so = 1.f / (1.f + __expf(-go));
  const float tg = tanhf(gg);
  const float c  = sf * cbuf[idx] + si * tg;
  cbuf[idx] = c;
  const float hv = so * tanhf(c);
  const __bf16 hh = (__bf16)hv;
  h_hi[idx] = hh;
  h_lo[idx] = (__bf16)(hv - (float)hh);
}

// ---------------------------------------------------------------------------
// Kernel 4: dec_part[b, k] = sum_h h[b,h] * W_hid[k, H + h]
// ---------------------------------------------------------------------------
__global__ __launch_bounds__(128) void dec_gemm_kernel(
    const __bf16* __restrict__ h_hi, const __bf16* __restrict__ h_lo,
    const __bf16* __restrict__ whid_hi, const __bf16* __restrict__ whid_lo,
    float* __restrict__ dec) {
  const int lane = threadIdx.x & 31;
  const int wave = threadIdx.x >> 5;
  const int n0 = (blockIdx.x * 16 + wave * 4) * 16;      // 4 of 64 tiles

  v8f acc[4];
  gemm_rowtile<4>(h_hi, h_lo, H_DIM, whid_hi, whid_lo, 2 * H_DIM, n0,
                  /*woff=*/H_DIM, H_DIM, lane, acc);

  const int n  = lane & 15;
  const int mb = (lane & 16) ? 8 : 0;
#pragma unroll
  for (int t = 0; t < 4; ++t) {
    float* out = dec + (size_t)mb * H_DIM + n0 + t * 16 + n;
#pragma unroll
    for (int r = 0; r < 8; ++r) out[(size_t)r * H_DIM] = acc[t][r];
  }
}

// ---------------------------------------------------------------------------
// Kernel 5: score[b,e] = sum_k relu(enc_part[b,e,k] + dec[b,k]) * W_out[k] + b_out
// One wave per (b,e); enc_part streamed from L2, dec/W_out staged in LDS.
// ---------------------------------------------------------------------------
__global__ __launch_bounds__(256) void score_kernel(
    const float* __restrict__ enc_part, const float* __restrict__ dec,
    const float* __restrict__ W_out, const float* __restrict__ b_out,
    float* __restrict__ out, int t) {
  __shared__ __align__(16) float dec_sh[H_DIM];
  __shared__ __align__(16) float wout_sh[H_DIM];
  const int b  = blockIdx.y;
  const int eg = blockIdx.x;
  for (int i = threadIdx.x; i < H_DIM; i += 256) {
    dec_sh[i]  = dec[(size_t)b * H_DIM + i];
    wout_sh[i] = W_out[i];
  }
  __syncthreads();

  const int lane = threadIdx.x & 31;
  const int wave = threadIdx.x >> 5;
  const int e = eg * 8 + wave;
  const float* row = enc_part + ((size_t)b * E_DIM + e) * H_DIM;

  float s = 0.f;
  for (int k = lane * 4; k < H_DIM; k += 128) {
    const float4 v = *(const float4*)(row + k);
    const float4 d = *(const float4*)(dec_sh + k);
    const float4 w = *(const float4*)(wout_sh + k);
    s += fmaxf(v.x + d.x, 0.f) * w.x;
    s += fmaxf(v.y + d.y, 0.f) * w.y;
    s += fmaxf(v.z + d.z, 0.f) * w.z;
    s += fmaxf(v.w + d.w, 0.f) * w.w;
  }
#pragma unroll
  for (int off = 16; off > 0; off >>= 1) s += __shfl_xor(s, off, 32);
  if (lane == 0)
    out[((size_t)b * DEC_STEPS + t) * E_DIM + e] = s + b_out[0];
}

// ---------------------------------------------------------------------------
// Kernel 6: masked softmax over E, in place on d_out (rows = B*DEC).
// ---------------------------------------------------------------------------
__global__ __launch_bounds__(256) void softmax_kernel(
    float* __restrict__ out, const unsigned char* __restrict__ mask) {
  __shared__ float red[256];
  const int row = blockIdx.x;      // = b*DEC + t
  const int b   = row / DEC_STEPS;
  float* s = out + (size_t)row * E_DIM;
  const unsigned char* m = mask + (size_t)b * E_DIM;

  float v[4];
  float mx = -INFINITY;
#pragma unroll
  for (int i = 0; i < 4; ++i) {
    const int e = threadIdx.x + i * 256;
    const float x = m[e] ? s[e] : -INFINITY;
    v[i] = x;
    mx = fmaxf(mx, x);
  }
  red[threadIdx.x] = mx;
  __syncthreads();
  for (int off = 128; off > 0; off >>= 1) {
    if (threadIdx.x < off)
      red[threadIdx.x] = fmaxf(red[threadIdx.x], red[threadIdx.x + off]);
    __syncthreads();
  }
  mx = red[0];
  __syncthreads();

  float sum = 0.f;
#pragma unroll
  for (int i = 0; i < 4; ++i) {
    v[i] = __expf(v[i] - mx);
    sum += v[i];
  }
  red[threadIdx.x] = sum;
  __syncthreads();
  for (int off = 128; off > 0; off >>= 1) {
    if (threadIdx.x < off) red[threadIdx.x] += red[threadIdx.x + off];
    __syncthreads();
  }
  const float inv = 1.f / red[0];
#pragma unroll
  for (int i = 0; i < 4; ++i) s[threadIdx.x + i * 256] = v[i] * inv;
}

// ---------------------------------------------------------------------------
extern "C" void kernel_launch(void* const* d_in, const int* in_sizes, int n_in,
                              void* d_out, int out_size, void* d_ws, size_t ws_size,
                              hipStream_t stream) {
  (void)in_sizes; (void)n_in; (void)out_size; (void)ws_size;

  const float*         enc   = (const float*)d_in[0];   // (B,E,H)
  const float*         h0    = (const float*)d_in[1];   // (1,B,H)
  const float*         c0    = (const float*)d_in[2];   // (1,B,H)
  const unsigned char* mask  = (const unsigned char*)d_in[3]; // (B,E) bool
  // d_in[4] = output_length (compile-time DEC_STEPS); d_in[5] = W_ih (unused)
  const float*         W_hh  = (const float*)d_in[6];   // (4H,H)
  const float*         b_ih  = (const float*)d_in[7];   // (4H,)
  const float*         b_hh  = (const float*)d_in[8];   // (4H,)
  const float*         W_hid = (const float*)d_in[9];   // (H,2H)
  const float*         b_hid = (const float*)d_in[10];  // (H,)
  const float*         W_out = (const float*)d_in[11];  // (1,H)
  const float*         b_out = (const float*)d_in[12];  // (1,)
  float*               out   = (float*)d_out;           // (B,DEC,E)

  // Workspace (~160 MB): enc_part | enc bf16 hi/lo | W_hid bf16 | W_hh bf16 |
  //                      h bf16 hi/lo | c | gates | dec_part
  char* ws = (char*)d_ws;
  size_t off = 0;
  const size_t ENC_N  = (size_t)B_DIM * E_DIM * H_DIM;  // 16.7M
  const size_t WHID_N = (size_t)H_DIM * 2 * H_DIM;      // 2.1M
  const size_t WHH_N  = (size_t)G_DIM * H_DIM;          // 4.2M
  const size_t HB_N   = (size_t)B_DIM * H_DIM;          // 16K

  float*  enc_part = (float*)(ws + off);  off += ENC_N * sizeof(float);
  __bf16* enc_hi   = (__bf16*)(ws + off); off += ENC_N * sizeof(__bf16);
  __bf16* enc_lo   = (__bf16*)(ws + off); off += ENC_N * sizeof(__bf16);
  __bf16* whid_hi  = (__bf16*)(ws + off); off += WHID_N * sizeof(__bf16);
  __bf16* whid_lo  = (__bf16*)(ws + off); off += WHID_N * sizeof(__bf16);
  __bf16* whh_hi   = (__bf16*)(ws + off); off += WHH_N * sizeof(__bf16);
  __bf16* whh_lo   = (__bf16*)(ws + off); off += WHH_N * sizeof(__bf16);
  __bf16* h_hi     = (__bf16*)(ws + off); off += HB_N * sizeof(__bf16);
  __bf16* h_lo     = (__bf16*)(ws + off); off += HB_N * sizeof(__bf16);
  float*  cbuf     = (float*)(ws + off);  off += HB_N * sizeof(float);
  float*  gates    = (float*)(ws + off);  off += (size_t)B_DIM * G_DIM * sizeof(float);
  float*  decb     = (float*)(ws + off);  off += HB_N * sizeof(float);

  hipMemcpyAsync(cbuf, c0, HB_N * sizeof(float), hipMemcpyDeviceToDevice, stream);

  // One-time bf16 hi/lo splits (4 f32/thread).
  split_bf16_kernel<<<ENC_N  / 4 / 256, 256, 0, stream>>>(enc,   enc_hi,  enc_lo);
  split_bf16_kernel<<<WHID_N / 4 / 256, 256, 0, stream>>>(W_hid, whid_hi, whid_lo);
  split_bf16_kernel<<<WHH_N  / 4 / 256, 256, 0, stream>>>(W_hh,  whh_hi,  whh_lo);
  split_bf16_kernel<<<HB_N   / 4 / 256, 256, 0, stream>>>(h0,    h_hi,    h_lo);

  // Big GEMM: 1024 M-tiles x 4 N-groups (block = 4 waves x 4 N-tiles).
  enc_gemm_kernel<<<dim3((B_DIM * E_DIM) / 16, (H_DIM / 16) / 16), 128, 0, stream>>>(
      enc_hi, enc_lo, whid_hi, whid_lo, b_hid, enc_part);

  for (int t = 0; t < DEC_STEPS; ++t) {
    gates_gemm_kernel<<<(G_DIM / 16) / 16, 128, 0, stream>>>(h_hi, h_lo, whh_hi,
                                                             whh_lo, b_ih, b_hh, gates);
    lstm_cell_kernel<<<(B_DIM * H_DIM) / 256, 256, 0, stream>>>(gates, cbuf, h_hi, h_lo);
    dec_gemm_kernel<<<(H_DIM / 16) / 16, 128, 0, stream>>>(h_hi, h_lo, whid_hi,
                                                           whid_lo, decb);
    score_kernel<<<dim3(E_DIM / 8, B_DIM), 256, 0, stream>>>(enc_part, decb, W_out,
                                                             b_out, out, t);
  }

  softmax_kernel<<<B_DIM * DEC_STEPS, 256, 0, stream>>>(out, mask);
}
